// Block_4346506904195
// MI455X (gfx1250) — matrix operand order repaired and checked by reading, not compiled, not run on hardware
//
#include <hip/hip_runtime.h>
#include <hip/hip_bf16.h>
#include <math.h>

// ---- problem constants -----------------------------------------------------
#define Bc   4
#define Sc   2048
#define Hc   16
#define HSc  64
#define Dc   1024
#define DFFc 4096
#define MRc  (Bc * Sc)          // 8192 rows
#define EPSc 1e-5f

typedef __bf16 bf16;
typedef __attribute__((ext_vector_type(16))) __bf16 v16bf;
typedef __attribute__((ext_vector_type(8)))  __bf16 v8bf;
typedef __attribute__((ext_vector_type(8)))  float  v8f;
typedef __attribute__((ext_vector_type(4)))  unsigned int v4u;
typedef __attribute__((ext_vector_type(8)))  int v8i;

static __device__ __forceinline__ v16bf cat8(v8bf lo, v8bf hi) {
  return __builtin_shufflevector(lo, hi, 0,1,2,3,4,5,6,7,8,9,10,11,12,13,14,15);
}
static __device__ __forceinline__ v8bf ld8(const bf16* p) {
  return *(const v8bf*)p;
}
static __device__ __forceinline__ void wave_lds_fence() {
#if __has_builtin(__builtin_amdgcn_fence)
  __builtin_amdgcn_fence(__ATOMIC_ACQ_REL, "wavefront");
#endif
}
static __device__ __forceinline__ unsigned int lds_off(const void* p) {
  return (unsigned int)(uintptr_t)p;     // LDS aperture: addr[31:0] is the LDS byte address
}

// ---- Tensor Data Mover: 2D tile load, global -> LDS, with LDS row padding --
// D# group0: count=1, lds_addr, global_addr[56:0], type=2
// D# group1: data_size=2B, pad_enable, pad_interval/amount, dims/strides
static __device__ __forceinline__ void tdm_tile_load(
    unsigned int lds_byte_addr, const bf16* gsrc,
    unsigned int tile_w, unsigned int tile_h, unsigned int stride_elems,
    unsigned int pad_int_code, unsigned int pad_amt_code) {
  const unsigned long long ga = (unsigned long long)(uintptr_t)gsrc;
  const unsigned int tensor_h = 1u << 20;        // generous (in-bounds tiles)
  v4u g0;
  g0[0] = 1u;                                    // count = 1 descriptor
  g0[1] = lds_byte_addr;
  g0[2] = (unsigned int)ga;
  g0[3] = ((unsigned int)(ga >> 32) & 0x01FFFFFFu) | 0x80000000u;  // type=2
  v8i g1;
  g1[0] = (int)((1u << 16) | (1u << 20) | (pad_int_code << 22) | (pad_amt_code << 25));
  g1[1] = (int)((stride_elems & 0xFFFFu) << 16);                   // tensor_dim0 lo16
  g1[2] = (int)(((stride_elems >> 16) & 0xFFFFu) |
                ((tensor_h & 0xFFFFu) << 16));                     // dim0 hi16 | dim1 lo16
  g1[3] = (int)(((tensor_h >> 16) & 0xFFFFu) | (tile_w << 16));    // dim1 hi16 | tile_dim0
  g1[4] = (int)(tile_h & 0xFFFFu);                                 // tile_dim1 (tile_dim2=0)
  g1[5] = (int)stride_elems;                                       // tensor_dim0_stride lo32
  g1[6] = 0;
  g1[7] = 0;
  asm volatile("tensor_load_to_lds %0, %1" :: "s"(g0), "s"(g1) : "memory");
}
static __device__ __forceinline__ void tensor_wait() {
#if __has_builtin(__builtin_amdgcn_s_wait_tensorcnt)
  __builtin_amdgcn_s_wait_tensorcnt(0);
#else
  asm volatile("s_wait_tensorcnt 0x0" ::: "memory");
#endif
}

// ---- weight conversion (fp32 -> bf16, TRANSPOSED to [N][K]) ----------------
// (H,D,HS) fp32 -> Wt[n][d], n = h*HS+hs
__global__ __launch_bounds__(256) void cvt_headw_t(const float* __restrict__ src,
                                                   bf16* __restrict__ dst) {
  int idx = blockIdx.x * 256 + threadIdx.x;      // N*K = 1M
  int n = idx >> 10, d = idx & 1023, h = n >> 6, k = n & 63;
  dst[idx] = (bf16)src[((size_t)h * Dc + d) * HSc + k];
}
// generic [K][N] fp32 -> [N][K] bf16
__global__ __launch_bounds__(256) void cvt_t(const float* __restrict__ src,
                                             bf16* __restrict__ dst, int Nn, int Kk) {
  int idx = blockIdx.x * 256 + threadIdx.x;
  int n = idx / Kk, k = idx % Kk;
  dst[idx] = (bf16)src[(size_t)k * Nn + n];
}

// ---- LayerNorm: fp32 row -> bf16 row ---------------------------------------
__global__ __launch_bounds__(256) void ln_bf16(const float* __restrict__ x,
                                               const float* __restrict__ w,
                                               const float* __restrict__ b,
                                               bf16* __restrict__ out) {
  __shared__ float red[16];
  const int row = blockIdx.x, tid = threadIdx.x;
  const int lane = tid & 31, wave = tid >> 5;
  const float4 v = ((const float4*)(x + (size_t)row * Dc))[tid];
  float s  = v.x + v.y + v.z + v.w;
  float s2 = v.x*v.x + v.y*v.y + v.z*v.z + v.w*v.w;
#pragma unroll
  for (int o = 16; o > 0; o >>= 1) { s += __shfl_xor(s, o, 32); s2 += __shfl_xor(s2, o, 32); }
  if (lane == 0) { red[wave] = s; red[8 + wave] = s2; }
  __syncthreads();
  float ts = 0.f, ts2 = 0.f;
#pragma unroll
  for (int i = 0; i < 8; ++i) { ts += red[i]; ts2 += red[8 + i]; }
  const float mu   = ts  * (1.0f / Dc);
  const float var  = ts2 * (1.0f / Dc) - mu * mu;
  const float rstd = rsqrtf(var + EPSc);
  const int c0 = tid * 4;
  struct alignas(8) bf4 { bf16 a[4]; } o4;
  const float xv[4] = { v.x, v.y, v.z, v.w };
#pragma unroll
  for (int i = 0; i < 4; ++i)
    o4.a[i] = (bf16)((xv[i] - mu) * rstd * w[c0 + i] + b[c0 + i]);
  *(bf4*)&out[(size_t)row * Dc + c0] = o4;
}

// ---- tiled WMMA GEMM: C[M,N] = A[M,K]bf16 @ Bt[N,K]bf16^T ------------------
// BM=BN=128, BK=32. 8 waves, wave tile 32x64. TDM double-buffered LDS.
#define EPI_QKV   0   // bf16 out scattered to [B,H,S,HS], +bias
#define EPI_RESID 1   // f32 out = acc + bias + resid
#define EPI_GELU  2   // bf16 out = gelu(acc + bias)
#define EPI_QKVT  3   // bf16 out scattered to [B,H,HS,S] (transposed V), +bias

template <int EPI>
__global__ __launch_bounds__(256)
void wmma_gemm(const bf16* __restrict__ A, const bf16* __restrict__ Bt,
               const float* __restrict__ bias, const float* __restrict__ resid,
               void* __restrict__ outp, int M, int N, int K) {
  __shared__ __align__(16) bf16 As[2][128 * 40];  // [m][k], pad 40 (TDM pads)
  __shared__ __align__(16) bf16 Bs[2][128 * 40];  // [n][k], pad 40 (TDM pads)
  const int tid    = threadIdx.x;
  const int lane   = tid & 31, wave = tid >> 5;
  const int laneHi = lane >> 4, lane16 = lane & 15;
  const int waveM  = (wave & 3) * 32;
  const int waveN  = (wave >> 2) * 64;
  const int m0 = blockIdx.y * 128;
  const int n0 = blockIdx.x * 128;
  const bf16* Abase = A  + (size_t)m0 * K;
  const bf16* Bbase = Bt + (size_t)n0 * K;

  v8f acc[2][4] = {};
  const int nk = K >> 5;

  if (wave == 0) {   // prologue: stage buffer 0 via Tensor Data Mover
    tdm_tile_load(lds_off(&As[0][0]), Abase, 32, 128, K, 3, 3);  // pad 4DW per 16DW -> stride 40
    tdm_tile_load(lds_off(&Bs[0][0]), Bbase, 32, 128, K, 3, 3);
  }
  for (int kt = 0; kt < nk; ++kt) {
    const int cur = kt & 1;
    if (wave == 0) tensor_wait();
    __syncthreads();                        // buf[cur] ready; buf[cur^1] free
    if (wave == 0 && kt + 1 < nk) {         // overlap DMA with WMMA
      tdm_tile_load(lds_off(&As[cur ^ 1][0]), Abase + (kt + 1) * 32, 32, 128, K, 3, 3);
      tdm_tile_load(lds_off(&Bs[cur ^ 1][0]), Bbase + (kt + 1) * 32, 32, 128, K, 3, 3);
    }
    v16bf afrag[2], bfrag[4];
#pragma unroll
    for (int mt = 0; mt < 2; ++mt) {
      const bf16* ap = &As[cur][(waveM + mt * 16 + lane16) * 40 + laneHi * 8];
      afrag[mt] = cat8(ld8(ap), ld8(ap + 16));
    }
#pragma unroll
    for (int nt = 0; nt < 4; ++nt) {
      const bf16* bp = &Bs[cur][(waveN + nt * 16 + lane16) * 40 + laneHi * 8];
      bfrag[nt] = cat8(ld8(bp), ld8(bp + 16));
    }
#pragma unroll
    for (int mt = 0; mt < 2; ++mt)
#pragma unroll
      for (int nt = 0; nt < 4; ++nt)
        acc[mt][nt] = __builtin_amdgcn_wmma_f32_16x16x32_bf16(
            false, afrag[mt], false, bfrag[nt], (short)0, acc[mt][nt], false, false);
  }

  // epilogue: C layout lane 0-15 -> (M=r, N=lane); lane 16-31 -> (M=r+8)
#pragma unroll
  for (int mt = 0; mt < 2; ++mt) {
#pragma unroll
    for (int nt = 0; nt < 4; ++nt) {
      const int n = n0 + waveN + nt * 16 + lane16;
      const float bv = bias[n];
#pragma unroll
      for (int r = 0; r < 8; ++r) {
        const int m = m0 + waveM + mt * 16 + r + laneHi * 8;
        float v = acc[mt][nt][r] + bv;
        if constexpr (EPI == EPI_QKV) {
          const int bb = m >> 11, s = m & (Sc - 1), h = n >> 6, hs = n & 63;
          ((bf16*)outp)[((size_t)(bb * Hc + h) * Sc + s) * HSc + hs] = (bf16)v;
        } else if constexpr (EPI == EPI_QKVT) {
          const int bb = m >> 11, s = m & (Sc - 1), h = n >> 6, hs = n & 63;
          ((bf16*)outp)[((size_t)(bb * Hc + h) * HSc + hs) * Sc + s] = (bf16)v;
        } else if constexpr (EPI == EPI_RESID) {
          ((float*)outp)[(size_t)m * N + n] = v + resid[(size_t)m * N + n];
        } else {
          ((bf16*)outp)[(size_t)m * N + n] =
              (bf16)(0.5f * v * (1.0f + erff(v * 0.70710678118654752f)));
        }
      }
    }
  }
}

// ---- flash attention: per (b,h), 64-query tile, 32-key tiles ---------------
// Q,K in [B,H,S,HS]; V in [B,H,HS,S] (pre-transposed by its GEMM epilogue).
__global__ __launch_bounds__(128)
void attn_kernel(const bf16* __restrict__ Q, const bf16* __restrict__ Kk,
                 const bf16* __restrict__ V, bf16* __restrict__ O) {
  __shared__ __align__(16) bf16 Qs[64 * 72];          // [q][hs]
  __shared__ __align__(16) bf16 Ks[2][32 * 72];       // [key][hs]
  __shared__ __align__(16) bf16 Vt[2][64 * 40];       // [hs][key]
  __shared__ __align__(16) bf16 Ps[4 * 16 * 40];      // per-wave P tile [row][key]

  const int tid    = threadIdx.x;
  const int lane   = tid & 31, wave = tid >> 5;
  const int laneHi = lane >> 4, lane16 = lane & 15;
  const int q0 = blockIdx.x * 64;
  const int bh = blockIdx.y;
  const int b = bh >> 4, h = bh & 15;
  const bf16* qp = Q  + (size_t)bh * Sc * HSc;
  const bf16* kp = Kk + (size_t)bh * Sc * HSc;
  const bf16* vp = V  + (size_t)bh * HSc * Sc;

  const int nkt = (q0 >> 5) + 2;                      // causal: keys <= q0+63
  if (wave == 0) {
    tdm_tile_load(lds_off(&Qs[0]),     qp + (size_t)q0 * HSc, 64, 64, HSc, 4, 3);  // stride 72
    tdm_tile_load(lds_off(&Ks[0][0]),  kp,                    64, 32, HSc, 4, 3);  // stride 72
    tdm_tile_load(lds_off(&Vt[0][0]),  vp,                    32, 64, Sc,  3, 3);  // stride 40
  }

  v8f o_acc[4] = {};
  float mrow[8], lrow[8];
#pragma unroll
  for (int r = 0; r < 8; ++r) { mrow[r] = -INFINITY; lrow[r] = 0.f; }

  for (int kt = 0; kt < nkt; ++kt) {
    const int cur = kt & 1;
    const int kbase = kt * 32;
    if (wave == 0) tensor_wait();
    __syncthreads();
    if (wave == 0 && kt + 1 < nkt) {
      tdm_tile_load(lds_off(&Ks[cur ^ 1][0]), kp + (size_t)(kbase + 32) * HSc, 64, 32, HSc, 4, 3);
      tdm_tile_load(lds_off(&Vt[cur ^ 1][0]), vp + (kbase + 32),               32, 64, Sc,  3, 3);
    }

    // S = Q @ K^T : 16x32 per wave, K-dim 64 -> 2 k-steps x 2 n-tiles
    v8f sacc[2] = {};
#pragma unroll
    for (int ks = 0; ks < 64; ks += 32) {
      const bf16* ap = &Qs[(wave * 16 + lane16) * 72 + ks + laneHi * 8];
      v16bf afrag = cat8(ld8(ap), ld8(ap + 16));
#pragma unroll
      for (int nt = 0; nt < 2; ++nt) {
        const bf16* bp = &Ks[cur][(nt * 16 + lane16) * 72 + ks + laneHi * 8];
        v16bf bfrag = cat8(ld8(bp), ld8(bp + 16));
        sacc[nt] = __builtin_amdgcn_wmma_f32_16x16x32_bf16(
            false, afrag, false, bfrag, (short)0, sacc[nt], false, false);
      }
    }

    // scale + causal mask + online softmax (half-wave owns 8 rows)
    float p[2][8];
#pragma unroll
    for (int r = 0; r < 8; ++r) {
      const int row = q0 + wave * 16 + r + laneHi * 8;
#pragma unroll
      for (int nt = 0; nt < 2; ++nt) {
        const int col = kbase + nt * 16 + lane16;
        const float sv = sacc[nt][r] * 0.125f;        // 1/sqrt(64)
        p[nt][r] = (col <= row) ? sv : -INFINITY;
      }
    }
#pragma unroll
    for (int r = 0; r < 8; ++r) {
      float mx = fmaxf(p[0][r], p[1][r]);
#pragma unroll
      for (int o = 1; o < 16; o <<= 1) mx = fmaxf(mx, __shfl_xor(mx, o, 32));
      const float mnew  = fmaxf(mrow[r], mx);
      const float scale = expf(mrow[r] - mnew);       // 0 on first tile
      float rs = 0.f;
#pragma unroll
      for (int nt = 0; nt < 2; ++nt) { p[nt][r] = expf(p[nt][r] - mnew); rs += p[nt][r]; }
#pragma unroll
      for (int o = 1; o < 16; o <<= 1) rs += __shfl_xor(rs, o, 32);
      lrow[r] = lrow[r] * scale + rs;
      mrow[r] = mnew;
#pragma unroll
      for (int nt = 0; nt < 4; ++nt) o_acc[nt][r] *= scale;
    }

    // C-layout P -> LDS (wave-private) -> A-layout fragment
    bf16* pw = &Ps[wave * 16 * 40];
#pragma unroll
    for (int r = 0; r < 8; ++r) {
      const int rowi = r + laneHi * 8;
#pragma unroll
      for (int nt = 0; nt < 2; ++nt)
        pw[rowi * 40 + nt * 16 + lane16] = (bf16)p[nt][r];
    }
    wave_lds_fence();
    {
      const bf16* ap = &pw[lane16 * 40 + laneHi * 8];
      v16bf pfrag = cat8(ld8(ap), ld8(ap + 16));
#pragma unroll
      for (int nt = 0; nt < 4; ++nt) {
        const bf16* bp = &Vt[cur][(nt * 16 + lane16) * 40 + laneHi * 8];
        v16bf vfrag = cat8(ld8(bp), ld8(bp + 16));
        o_acc[nt] = __builtin_amdgcn_wmma_f32_16x16x32_bf16(
            false, pfrag, false, vfrag, (short)0, o_acc[nt], false, false);
      }
    }
  }

  // write O as bf16 to [B,S, h*HS+hs] (rows ready for proj GEMM)
#pragma unroll
  for (int nt = 0; nt < 4; ++nt) {
    const int col = h * HSc + nt * 16 + lane16;
#pragma unroll
    for (int r = 0; r < 8; ++r) {
      const int row = q0 + wave * 16 + r + laneHi * 8;
      O[(size_t)(b * Sc + row) * Dc + col] = (bf16)(o_acc[nt][r] / lrow[r]);
    }
  }
}

// ---- orchestration ---------------------------------------------------------
extern "C" void kernel_launch(void* const* d_in, const int* in_sizes, int n_in,
                              void* d_out, int out_size, void* d_ws, size_t ws_size,
                              hipStream_t stream) {
  (void)in_sizes; (void)n_in; (void)out_size; (void)ws_size;
  const float* x    = (const float*)d_in[0];
  const float* Wq   = (const float*)d_in[1];
  const float* bq   = (const float*)d_in[2];
  const float* Wk   = (const float*)d_in[3];
  const float* bk   = (const float*)d_in[4];
  const float* Wv   = (const float*)d_in[5];
  const float* bv   = (const float*)d_in[6];
  const float* Wp   = (const float*)d_in[7];
  const float* bp   = (const float*)d_in[8];
  const float* ln1w = (const float*)d_in[9];
  const float* ln1b = (const float*)d_in[10];
  const float* ln2w = (const float*)d_in[11];
  const float* ln2b = (const float*)d_in[12];
  const float* W1   = (const float*)d_in[13];
  const float* b1   = (const float*)d_in[14];
  const float* W2   = (const float*)d_in[15];
  const float* b2   = (const float*)d_in[16];

  char* ws = (char*)d_ws;
  size_t off = 0;
  auto alloc = [&](size_t bytes) {
    char* p = ws + off; off += (bytes + 255) & ~(size_t)255; return p;
  };
  bf16* wq_b = (bf16*)alloc((size_t)Dc * Dc * 2);    // Wt layouts: [N][K]
  bf16* wk_b = (bf16*)alloc((size_t)Dc * Dc * 2);
  bf16* wv_b = (bf16*)alloc((size_t)Dc * Dc * 2);
  bf16* wp_b = (bf16*)alloc((size_t)Dc * Dc * 2);
  bf16* w1_b = (bf16*)alloc((size_t)Dc * DFFc * 2);
  bf16* w2_b = (bf16*)alloc((size_t)Dc * DFFc * 2);
  bf16* xn   = (bf16*)alloc((size_t)MRc * Dc * 2);
  bf16* qb   = (bf16*)alloc((size_t)MRc * Dc * 2);
  bf16* kb   = (bf16*)alloc((size_t)MRc * Dc * 2);
  bf16* vb   = (bf16*)alloc((size_t)MRc * Dc * 2);   // [B,H,HS,S]
  bf16* ao   = (bf16*)alloc((size_t)MRc * Dc * 2);
  float* x2  = (float*)alloc((size_t)MRc * Dc * 4);
  bf16* xn2  = (bf16*)alloc((size_t)MRc * Dc * 2);
  bf16* hh   = (bf16*)alloc((size_t)MRc * DFFc * 2);

  // weight conversions (fp32 -> bf16, transposed to [N][K])
  cvt_headw_t<<<(Dc * Hc * HSc) / 256, 256, 0, stream>>>(Wq, wq_b);
  cvt_headw_t<<<(Dc * Hc * HSc) / 256, 256, 0, stream>>>(Wk, wk_b);
  cvt_headw_t<<<(Dc * Hc * HSc) / 256, 256, 0, stream>>>(Wv, wv_b);
  cvt_t<<<(Dc * Dc)   / 256, 256, 0, stream>>>(Wp, wp_b, Dc,   Dc);
  cvt_t<<<(Dc * DFFc) / 256, 256, 0, stream>>>(W1, w1_b, DFFc, Dc);   // -> [DFF][D]
  cvt_t<<<(Dc * DFFc) / 256, 256, 0, stream>>>(W2, w2_b, Dc,   DFFc); // -> [D][DFF]

  // LN1
  ln_bf16<<<MRc, 256, 0, stream>>>(x, ln1w, ln1b, xn);

  // QKV projections (V transposed to [B,H,HS,S])
  dim3 gD(Dc / 128, MRc / 128);
  wmma_gemm<EPI_QKV ><<<gD, 256, 0, stream>>>(xn, wq_b, bq, nullptr, qb, MRc, Dc, Dc);
  wmma_gemm<EPI_QKV ><<<gD, 256, 0, stream>>>(xn, wk_b, bk, nullptr, kb, MRc, Dc, Dc);
  wmma_gemm<EPI_QKVT><<<gD, 256, 0, stream>>>(xn, wv_b, bv, nullptr, vb, MRc, Dc, Dc);

  // attention
  attn_kernel<<<dim3(Sc / 64, Bc * Hc), 128, 0, stream>>>(qb, kb, vb, ao);

  // output projection + residual -> x2 (f32)
  wmma_gemm<EPI_RESID><<<gD, 256, 0, stream>>>(ao, wp_b, bp, x, x2, MRc, Dc, Dc);

  // LN2
  ln_bf16<<<MRc, 256, 0, stream>>>(x2, ln2w, ln2b, xn2);

  // FFN
  dim3 gF(DFFc / 128, MRc / 128);
  wmma_gemm<EPI_GELU ><<<gF, 256, 0, stream>>>(xn2, w1_b, b1, nullptr, hh, MRc, DFFc, Dc);
  wmma_gemm<EPI_RESID><<<gD, 256, 0, stream>>>(hh, w2_b, b2, x2, (float*)d_out, MRc, Dc, DFFc);
}
